// QLSTMPlus_65481071397478
// MI455X (gfx1250) — compile-verified, matrix operand order checked by reading, not run on hardware
//
#include <hip/hip_runtime.h>
#include <math.h>

typedef __bf16 bf16_t;
typedef __attribute__((ext_vector_type(8)))  bf16_t v8bf;
typedef __attribute__((ext_vector_type(16))) bf16_t v16bf;
typedef __attribute__((ext_vector_type(8)))  float  v8f;

// Problem constants
#define S_LEN 512
#define B_DIM 64
#define D_DIM 512
#define H_DIM 1024
#define K4    1536            // D + H  (fused [x;h] @ W4)
#define N4    4096            // 4*H
#define KT_N  48              // K4 / 32
#define A_STRIDE 1544         // 1536 + 8 bf16 pad (bank-conflict avoidance)
#define Z_STRIDE 68           // 64 + 4 f32 pad
#define NBLOCKS  64           // persistent grid: 16 h-tiles x 4 b-tiles

// Workspace byte offsets
#define OFF_W4   0u                       // 1536*4096*2  = 12,582,912
#define OFF_XBF  12582912u                // 512*64*512*2 = 33,554,432
#define OFF_B4   46137344u                // 4096*4       = 16,384
#define OFF_HPP  46153728u                // 2*64*1024*2  = 262,144
#define OFF_CNT  46415872u                // 512*4        = 2,048

// ---------------- one-time setup kernels ----------------

__global__ void conv_x_kernel(const float* __restrict__ x, bf16_t* __restrict__ xbf, int n) {
    for (int i = blockIdx.x * blockDim.x + threadIdx.x; i < n; i += gridDim.x * blockDim.x)
        xbf[i] = (bf16_t)x[i];
}

// Pack W4 = concat(Wf,Wi,Wg,Wo, axis=1) (1536 x 4096 f32) into the WMMA
// B-operand swizzle: tile (nt,kt) = 16 cols x 32 K, stored as 32 lanes x 16 bf16
// contiguous (lane L<16: N=nt*16+L, K = {0..7,16..23}; lane L>=16: same N, K={8..15,24..31}).
// Tiles for one nt are contiguous over kt so the K-loop streams 1 KB/iter.
__global__ void pack_w_kernel(const float* __restrict__ Wf, const float* __restrict__ Wi,
                              const float* __restrict__ Wg, const float* __restrict__ Wo,
                              bf16_t* __restrict__ w4) {
    const int n = K4 * N4;
    for (int idx = blockIdx.x * blockDim.x + threadIdx.x; idx < n; idx += gridDim.x * blockDim.x) {
        int tile   = idx >> 9;          // /512 elements per tile
        int within = idx & 511;
        int lane   = within >> 4;
        int e      = within & 15;
        int nt = tile / KT_N;
        int kt = tile - nt * KT_N;
        int ncol = (nt << 4) + (lane & 15);            // global column in [0,4096)
        int k    = (kt << 5) + ((lane >> 4) << 3) + ((e >> 3) << 4) + (e & 7);
        int g    = ncol >> 10;
        int nin  = ncol & 1023;
        const float* W = (g == 0) ? Wf : (g == 1) ? Wi : (g == 2) ? Wg : Wo;
        w4[idx] = (bf16_t)W[(size_t)k * H_DIM + nin];
    }
}

__global__ void init_misc_kernel(const float* __restrict__ bf_, const float* __restrict__ bi_,
                                 const float* __restrict__ bg_, const float* __restrict__ bo_,
                                 float* __restrict__ b4, bf16_t* __restrict__ h0buf,
                                 int* __restrict__ cnt) {
    int i = blockIdx.x * blockDim.x + threadIdx.x;    // 65536 threads
    if (i < N4) {
        int g = i >> 10;
        const float* b = (g == 0) ? bf_ : (g == 1) ? bi_ : (g == 2) ? bg_ : bo_;
        b4[i] = b[i & 1023];
    }
    if (i < S_LEN) cnt[i] = 0;
    if (i < B_DIM * H_DIM) h0buf[i] = (bf16_t)0.0f;
}

// ---------------- persistent recurrent kernel ----------------
// grid = (16, 4): blockIdx.x -> 64-wide H tile, blockIdx.y -> 16-row batch tile.
// 8 waves; wave w -> gate (w>>1), column subtiles 2*(w&1), 2*(w&1)+1.
// Whole 512-step scan runs in ONE launch; steps separated by a device-wide
// atomic arrive/spin barrier. Cell state c lives in VGPRs for all 512 steps.

__device__ __forceinline__ float sigmoid_f(float x) {
    return 1.0f / (1.0f + __expf(-x));
}

__device__ __forceinline__ void async_copy_b128_to_lds(const void* gsrc, uint32_t lds_byte_off) {
    // GLOBAL_LOAD_ASYNC_TO_LDS_B128: copies 16 B from global memory into LDS at the given byte offset; tracked by ASYNCcnt.
    asm volatile("global_load_async_to_lds_b128 %0, %1, off"
                 :
                 : "v"(lds_byte_off), "v"((unsigned long long)(uintptr_t)gsrc)
                 : "memory");
}

__device__ __forceinline__ void prefetch_wgp(const void* p, int byte_off) {
    // WGP-scope (scope 0) prefetch: pull into all cache levels, stop on WGP hit.
    // Immediate offset keeps address math folded into the instruction.
    asm volatile("global_prefetch_b8 %0, off offset:%1"
                 :
                 : "v"((unsigned long long)(uintptr_t)p), "i"(byte_off));
}

__global__ __launch_bounds__(256) void lstm_persist_kernel(
    const bf16_t* __restrict__ xbf, bf16_t* __restrict__ hpp,
    int* __restrict__ cnt,
    const bf16_t* __restrict__ w4, const float* __restrict__ b4,
    float* __restrict__ out)
{
    // A region (16 x 1544 bf16 = 49,408 B) reused as Z region (4*16*68 f32) per step.
    __shared__ alignas(16) char smem[16 * A_STRIDE * 2];
    bf16_t* Alds = (bf16_t*)smem;
    float*  Zlds = (float*)smem;

    const int tid = threadIdx.x;
    const int h0  = blockIdx.x * 64;
    const int b0  = blockIdx.y * 16;

    // ---- loop-invariant tile setup ----
    const int wv   = tid >> 5;
    const int lane = tid & 31;
    const int gate = wv >> 1;
    const int half = wv & 1;
    const int j0   = half * 2;
    const int nt0  = gate * 64 + blockIdx.x * 4 + j0;
    const bf16_t* pB0_base = w4 + (size_t)nt0 * (KT_N * 512) + (size_t)lane * 16;
    const bf16_t* pB1_base = pB0_base + (size_t)(KT_N * 512);
    const int arow  = (lane & 15) * A_STRIDE;
    const int khalf = (lane >> 4) << 3;               // 0 or 8
    const int nl    = lane & 15;
    const int mhi   = (lane >> 4) << 3;
    const float bias0 = b4[gate * H_DIM + h0 + j0 * 16 + nl];
    const float bias1 = b4[gate * H_DIM + h0 + (j0 + 1) * 16 + nl];

    float c_reg[4] = {0.f, 0.f, 0.f, 0.f};            // cell state, VGPR-resident

    for (int s = 0; s < S_LEN; ++s) {
        const bf16_t* hin  = hpp + (size_t)(s & 1) * (B_DIM * H_DIM);
        bf16_t*       hout = hpp + (size_t)((s + 1) & 1) * (B_DIM * H_DIM);

        // ---- stage A = [x_s ; h] rows b0..b0+15 into LDS via async-to-LDS ----
        const bf16_t* xs = xbf + (size_t)s * (B_DIM * D_DIM);
        #pragma unroll
        for (int i = 0; i < 12; ++i) {                // 16 rows * 192 chunks / 256 threads
            int q   = tid + 256 * i;
            int row = q / 192;
            int cq  = q - row * 192;
            int col = cq * 8;
            const bf16_t* src = (cq < 64)
                ? (xs  + (size_t)(b0 + row) * D_DIM + col)
                : (hin + (size_t)(b0 + row) * H_DIM + (col - D_DIM));
            uint32_t ldsb = (uint32_t)(uintptr_t)(&Alds[row * A_STRIDE + col]);
            async_copy_b128_to_lds(src, ldsb);
        }
        asm volatile("s_wait_asynccnt 0x0" ::: "memory");
        __syncthreads();

        // ---- K-loop: z = [x;h] @ W4 via bf16 WMMA, f32 accumulate ----
        const bf16_t* pB0 = pB0_base;
        const bf16_t* pB1 = pB1_base;
        v8f acc0 = {};
        v8f acc1 = {};
        for (int ktg = 0; ktg < KT_N; ktg += 4) {
            // Prefetch next group's 4 KB per weight stream from the group base
            // register with immediate offsets (no extra address VALU).
            prefetch_wgp(pB0, 0x1000);
            prefetch_wgp(pB0, 0x1400);
            prefetch_wgp(pB0, 0x1800);
            prefetch_wgp(pB0, 0x1c00);
            prefetch_wgp(pB1, 0x1000);
            prefetch_wgp(pB1, 0x1400);
            prefetch_wgp(pB1, 0x1800);
            prefetch_wgp(pB1, 0x1c00);
            #pragma unroll
            for (int ku = 0; ku < 4; ++ku) {
                const int kb = (ktg + ku) * 32 + khalf;
                v8bf alo = *(const v8bf*)(&Alds[arow + kb]);
                v8bf ahi = *(const v8bf*)(&Alds[arow + kb + 16]);
                v16bf a = __builtin_shufflevector(alo, ahi,
                            0, 1, 2, 3, 4, 5, 6, 7, 8, 9, 10, 11, 12, 13, 14, 15);
                v16bf bm0 = *(const v16bf*)(pB0 + ku * 512);
                v16bf bm1 = *(const v16bf*)(pB1 + ku * 512);
                acc0 = __builtin_amdgcn_wmma_f32_16x16x32_bf16(false, a, false, bm0,
                                                               (short)0, acc0, false, false);
                acc1 = __builtin_amdgcn_wmma_f32_16x16x32_bf16(false, a, false, bm1,
                                                               (short)0, acc1, false, false);
            }
            pB0 += 4 * 512;
            pB1 += 4 * 512;
        }

        __syncthreads();   // all waves done reading Alds; region becomes Zlds

        // ---- spill gate tiles (+bias) to LDS for cross-wave elementwise ----
        {
            float* zg = &Zlds[gate * (16 * Z_STRIDE)];
            #pragma unroll
            for (int r = 0; r < 8; ++r) {
                const int row = r + mhi;
                zg[row * Z_STRIDE + j0 * 16 + nl]       = acc0[r] + bias0;
                zg[row * Z_STRIDE + (j0 + 1) * 16 + nl] = acc1[r] + bias1;
            }
        }
        __syncthreads();

        // ---- gates + cell update: 1024 elements, 4 per thread; c in VGPRs ----
        #pragma unroll
        for (int e = 0; e < 4; ++e) {
            const int elem = tid + 256 * e;
            const int col  = elem & 63;
            const int row  = elem >> 6;
            const float zf = Zlds[0 * (16 * Z_STRIDE) + row * Z_STRIDE + col];
            const float zi = Zlds[1 * (16 * Z_STRIDE) + row * Z_STRIDE + col];
            const float zg = Zlds[2 * (16 * Z_STRIDE) + row * Z_STRIDE + col];
            const float zo = Zlds[3 * (16 * Z_STRIDE) + row * Z_STRIDE + col];
            const float fv = sigmoid_f(zf);
            const float iv = sigmoid_f(zi);
            const float gv = tanhf(zg);
            const float ov = sigmoid_f(zo);
            const float cn = fv * c_reg[e] + iv * gv;
            c_reg[e] = cn;
            const float hn = ov * tanhf(cn);
            const size_t gidx = (size_t)(b0 + row) * H_DIM + (h0 + col);
            out[(size_t)s * (B_DIM * H_DIM) + gidx] = hn;
            hout[gidx] = (bf16_t)hn;
            if (s == S_LEN - 1) {
                out[(size_t)S_LEN * (B_DIM * H_DIM) + gidx]                   = hn;  // hx
                out[(size_t)S_LEN * (B_DIM * H_DIM) + (B_DIM * H_DIM) + gidx] = cn;  // cx
            }
        }

        // ---- device-wide step barrier (release h, arrive, spin, acquire) ----
        __threadfence();              // each thread drains its h stores to device scope
        __syncthreads();
        if (tid == 0) {
            __hip_atomic_fetch_add(&cnt[s], 1, __ATOMIC_ACQ_REL, __HIP_MEMORY_SCOPE_AGENT);
            while (__hip_atomic_load(&cnt[s], __ATOMIC_ACQUIRE, __HIP_MEMORY_SCOPE_AGENT)
                   < NBLOCKS) {
                __builtin_amdgcn_s_sleep(1);
            }
        }
        __syncthreads();              // whole block held until barrier round s completes
        __threadfence();              // invalidate WGP$ before reading next-step h
    }
}

// ---------------- launcher ----------------

extern "C" void kernel_launch(void* const* d_in, const int* in_sizes, int n_in,
                              void* d_out, int out_size, void* d_ws, size_t ws_size,
                              hipStream_t stream) {
    (void)in_sizes; (void)n_in; (void)out_size; (void)ws_size;

    const float* x   = (const float*)d_in[0];
    const float* Wf  = (const float*)d_in[1];
    const float* bfv = (const float*)d_in[2];
    const float* Wi  = (const float*)d_in[3];
    const float* biv = (const float*)d_in[4];
    const float* Wg  = (const float*)d_in[5];
    const float* bgv = (const float*)d_in[6];
    const float* Wo  = (const float*)d_in[7];
    const float* bov = (const float*)d_in[8];
    float* out = (float*)d_out;

    char* ws = (char*)d_ws;
    bf16_t* w4    = (bf16_t*)(ws + OFF_W4);
    bf16_t* xbf   = (bf16_t*)(ws + OFF_XBF);
    float*  b4    = (float*)(ws + OFF_B4);
    bf16_t* hping = (bf16_t*)(ws + OFF_HPP);
    int*    cnt   = (int*)(ws + OFF_CNT);

    // one-time prep (re-run every call for determinism)
    conv_x_kernel<<<8192, 256, 0, stream>>>(x, xbf, S_LEN * B_DIM * D_DIM);
    pack_w_kernel<<<8192, 256, 0, stream>>>(Wf, Wi, Wg, Wo, w4);
    init_misc_kernel<<<256, 256, 0, stream>>>(bfv, biv, bgv, bov, b4, hping, cnt);

    // entire 512-step recurrent scan: ONE persistent launch, device-wide
    // atomic barriers between steps (64 blocks are always co-resident).
    dim3 grid(16, 4);
    dim3 block(256);
    lstm_persist_kernel<<<grid, block, 0, stream>>>(xbf, hping, cnt, w4, b4, out);
}
